// GTN_17162689314910
// MI455X (gfx1250) — matrix operand order repaired and checked by reading, not compiled, not run on hardware
//
#include <hip/hip_runtime.h>
#include <hip/hip_bf16.h>

// ---------------------------------------------------------------------------
// GTN forward on MI455X (gfx1250). Compute-bound on 4x 2048^3 GEMMs
// (~69 GFLOP); everything fits in the 192MB L2 (HBM time ~15us), so the
// matmuls go through v_wmma_f32_16x16x32_bf16 (bf16 in / f32 accumulate),
// with double-buffered LDS tiles fed by global_load_async_to_lds_b128
// (ASYNCcnt) when the toolchain exposes the builtin.
// ---------------------------------------------------------------------------

typedef __bf16 bf16;
typedef __attribute__((ext_vector_type(16))) __bf16 v16bf;
typedef __attribute__((ext_vector_type(8)))  float  v8f;
typedef __attribute__((ext_vector_type(4)))  int    v4i;

union FragAB { v16bf v; uint4 q[2]; };
union BF4    { bf16 h[4]; uint2 u; };

static constexpr int    Ndim = 2048;
static constexpr size_t NN   = (size_t)Ndim * Ndim;
static constexpr int    Cch  = 2;
static constexpr int    Ech  = 5;
static constexpr int    WIN  = 256;
static constexpr int    WOUT = 64;
static constexpr int    LDS_STR = 48;   // 32 K + pad; 96B row stride keeps 16B align

#if __has_builtin(__builtin_amdgcn_global_load_async_to_lds_b128)
#define ASYNC_LDS 1
typedef __attribute__((address_space(1))) v4i gv4i;   // global int4 (AS1)
typedef __attribute__((address_space(3))) v4i lv4i;   // LDS int4 (AS3)
#else
#define ASYNC_LDS 0
#endif

__device__ __forceinline__ void wait_async() {
#if ASYNC_LDS
#if __has_builtin(__builtin_amdgcn_s_wait_asynccnt)
  __builtin_amdgcn_s_wait_asynccnt(0);
#else
  asm volatile("s_wait_asynccnt 0" ::: "memory");
#endif
#endif
}

__device__ __forceinline__ void async_copy16(const bf16* gp, bf16* sp) {
#if ASYNC_LDS
  __builtin_amdgcn_global_load_async_to_lds_b128((gv4i*)gp, (lv4i*)sp, 0, 0);
#else
  *(uint4*)sp = *(const uint4*)gp;
#endif
}

// Copy one 128x32 bf16 tile (row-major, row stride ld) into LDS (stride 48).
// 512 16B chunks, 2 per thread; async (ASYNCcnt) when available.
__device__ __forceinline__ void stage128x32(const bf16* __restrict__ g, int ld,
                                            bf16* s, int tid) {
#pragma unroll
  for (int ch = 0; ch < 2; ++ch) {
    int cidx = tid + ch * 256;
    int row  = cidx >> 2;
    int off  = (cidx & 3) << 3;
    const bf16* gp = g + (size_t)row * ld + off;
    bf16* sp = s + row * LDS_STR + off;
    async_copy16(gp, sp);
    __builtin_prefetch(gp + 64, 0, 1);   // speculative: next K tiles
  }
}

// ---------------------------------------------------------------------------
__device__ __forceinline__ void softmax5(const float* __restrict__ w, float s[Cch][Ech]) {
#pragma unroll
  for (int c = 0; c < Cch; ++c) {
    float m = w[c * Ech];
#pragma unroll
    for (int e = 1; e < Ech; ++e) m = fmaxf(m, w[c * Ech + e]);
    float sum = 0.f;
#pragma unroll
    for (int e = 0; e < Ech; ++e) { s[c][e] = __expf(w[c * Ech + e] - m); sum += s[c][e]; }
    float inv = 1.f / sum;
#pragma unroll
    for (int e = 0; e < Ech; ++e) s[c][e] *= inv;
  }
}

// ---------------------------------------------------------------------------
// Fused GTConv over 32x32 tiles: one pass over A[E,N,N] emits
//   a  [C,N,N]  row-major   (GEMM1 A operand)
//   bT [C,N,N]  transposed  (GEMM1 B operand, pre-transposed for staging)
//   a1T[C,N,N]  transposed  (GEMM2 B operand)
// ---------------------------------------------------------------------------
__global__ __launch_bounds__(256) void k_gtconv(
    const float* __restrict__ A,
    const float* __restrict__ w1, const float* __restrict__ w2, const float* __restrict__ w3,
    bf16* __restrict__ a_out, bf16* __restrict__ bT_out, bf16* __restrict__ a1T_out)
{
  __shared__ bf16 tb[Cch][32][36];
  __shared__ bf16 t1[Cch][32][36];
  const int j0 = blockIdx.x * 32;
  const int i0 = blockIdx.y * 32;
  const int t  = threadIdx.x;
  const int r  = t >> 3;         // 0..31 tile row
  const int c4 = (t & 7) << 2;   // 0..28 tile col (4-wide)

  float s1[Cch][Ech], s2[Cch][Ech], s3[Cch][Ech];
  softmax5(w1, s1); softmax5(w2, s2); softmax5(w3, s3);

  float vv[Ech][4];
  const float* base = A + (size_t)(i0 + r) * Ndim + j0 + c4;
#pragma unroll
  for (int e = 0; e < Ech; ++e) {
    float4 f = *(const float4*)(base + (size_t)e * NN);
    vv[e][0] = f.x; vv[e][1] = f.y; vv[e][2] = f.z; vv[e][3] = f.w;
  }
#pragma unroll
  for (int c = 0; c < Cch; ++c) {
    float sa[4] = {0,0,0,0}, sb[4] = {0,0,0,0}, sc[4] = {0,0,0,0};
#pragma unroll
    for (int e = 0; e < Ech; ++e)
#pragma unroll
      for (int k = 0; k < 4; ++k) {
        sa[k] += s1[c][e] * vv[e][k];
        sb[k] += s2[c][e] * vv[e][k];
        sc[k] += s3[c][e] * vv[e][k];
      }
    BF4 pa;
#pragma unroll
    for (int k = 0; k < 4; ++k) {
      pa.h[k] = (bf16)sa[k];
      tb[c][r][c4 + k] = (bf16)sb[k];
      t1[c][r][c4 + k] = (bf16)sc[k];
    }
    *(uint2*)(a_out + (size_t)c * NN + (size_t)(i0 + r) * Ndim + j0 + c4) = pa.u;
  }
  __syncthreads();
  // transposed emit: output row j0+r, cols i0+c4..+3
#pragma unroll
  for (int c = 0; c < Cch; ++c) {
    BF4 pb, p1;
#pragma unroll
    for (int k = 0; k < 4; ++k) {
      pb.h[k] = tb[c][c4 + k][r];
      p1.h[k] = t1[c][c4 + k][r];
    }
    size_t o = (size_t)c * NN + (size_t)(j0 + r) * Ndim + i0 + c4;
    *(uint2*)(bT_out  + o) = pb.u;
    *(uint2*)(a1T_out + o) = p1.u;
  }
}

// ---------------------------------------------------------------------------
// bf16 GEMM: C[c](MxN,f32) = A[c](MxK,row) * BT[c](NxK,row)^T, f32 accumulate.
// 128x128 block, BK=32, 256 threads = 8 waves, wave tile 64x32 (4x2 frags),
// double-buffered LDS with async global->LDS staging.
// ---------------------------------------------------------------------------
__global__ __launch_bounds__(256) void k_gemm_bf16(
    const bf16* __restrict__ Ag, const bf16* __restrict__ BTg, float* __restrict__ Cg,
    int M, int Nn, int K)
{
  const int c = blockIdx.z;
  Ag  += (size_t)c * M * K;
  BTg += (size_t)c * Nn * K;
  Cg  += (size_t)c * M * Nn;
  const int n0 = blockIdx.x * 128;
  const int m0 = blockIdx.y * 128;

  __shared__ bf16 sA[2][128 * LDS_STR];
  __shared__ bf16 sB[2][128 * LDS_STR];   // sB[n][k]

  const int tid  = threadIdx.x;
  const int wave = tid >> 5, lane = tid & 31;
  const int lmod = lane & 15, lhi = lane >> 4;
  const int wm = wave & 1;   // 2 waves over M
  const int wn = wave >> 1;  // 4 waves over N

  v8f acc[4][2];
#pragma unroll
  for (int mi = 0; mi < 4; ++mi)
#pragma unroll
    for (int ni = 0; ni < 2; ++ni) acc[mi][ni] = (v8f){0.f,0.f,0.f,0.f,0.f,0.f,0.f,0.f};

  const bf16* Abase = Ag  + (size_t)m0 * K;
  const bf16* Bbase = BTg + (size_t)n0 * K;

  stage128x32(Abase, K, sA[0], tid);
  stage128x32(Bbase, K, sB[0], tid);
  wait_async();
  __syncthreads();

  const int nT = K >> 5;
  for (int t = 0; t < nT; ++t) {
    const int cur = t & 1;
    if (t + 1 < nT) {
      stage128x32(Abase + (t + 1) * 32, K, sA[cur ^ 1], tid);
      stage128x32(Bbase + (t + 1) * 32, K, sB[cur ^ 1], tid);
    }
    FragAB aF[4], bF[2];
#pragma unroll
    for (int mi = 0; mi < 4; ++mi) {
      const bf16* p = &sA[cur][(wm * 64 + mi * 16 + lmod) * LDS_STR + lhi * 8];
      aF[mi].q[0] = *(const uint4*)p;          // K = lhi*8 .. +7
      aF[mi].q[1] = *(const uint4*)(p + 16);   // K = 16+lhi*8 .. +7
    }
#pragma unroll
    for (int ni = 0; ni < 2; ++ni) {
      const bf16* p = &sB[cur][(wn * 32 + ni * 16 + lmod) * LDS_STR + lhi * 8];
      bF[ni].q[0] = *(const uint4*)p;          // K = lhi*8 .. +7
      bF[ni].q[1] = *(const uint4*)(p + 16);   // K = 16+lhi*8 .. +7
    }
#pragma unroll
    for (int mi = 0; mi < 4; ++mi)
#pragma unroll
      for (int ni = 0; ni < 2; ++ni)
        acc[mi][ni] = __builtin_amdgcn_wmma_f32_16x16x32_bf16(
            false, aF[mi].v, false, bF[ni].v, (short)0, acc[mi][ni], false, false);
    wait_async();
    __syncthreads();
  }

  // C/D layout: VGPR v -> M = v (+8 for upper half-lanes), N = lane%16
#pragma unroll
  for (int mi = 0; mi < 4; ++mi)
#pragma unroll
    for (int ni = 0; ni < 2; ++ni) {
      int col = n0 + wn * 32 + ni * 16 + lmod;
#pragma unroll
      for (int v = 0; v < 8; ++v) {
        int row = m0 + wm * 64 + mi * 16 + lhi * 8 + v;
        Cg[(size_t)row * Nn + col] = acc[mi][ni][v];
      }
    }
}

// ---------------------------------------------------------------------------
// deg[j] = (add?1:0) + sum_{i!=j} H[i][j]; store 1/(deg+eps) (0 if deg==0).
// float4-vectorized: each thread owns 4 columns.
// ---------------------------------------------------------------------------
__global__ __launch_bounds__(256) void k_colsum(
    const float* __restrict__ H, float* __restrict__ dinv, int addone)
{
  int c  = blockIdx.y;
  int j4 = (blockIdx.x * 256 + threadIdx.x) << 2;
  const float* Hc = H + (size_t)c * NN;
  float s[4];
  float init = addone ? 1.f : 0.f;
#pragma unroll
  for (int e = 0; e < 4; ++e) s[e] = init;
  for (int i = 0; i < Ndim; ++i) {
    float4 v = *(const float4*)(Hc + (size_t)i * Ndim + j4);
    float x[4] = {v.x, v.y, v.z, v.w};
#pragma unroll
    for (int e = 0; e < 4; ++e) s[e] += (i == j4 + e) ? 0.f : x[e];
  }
  float4 o;
  o.x = (s[0] == 0.f) ? 0.f : 1.f / (s[0] + 1e-8f);
  o.y = (s[1] == 0.f) ? 0.f : 1.f / (s[1] + 1e-8f);
  o.z = (s[2] == 0.f) ? 0.f : 1.f / (s[2] + 1e-8f);
  o.w = (s[3] == 0.f) ? 0.f : 1.f / (s[3] + 1e-8f);
  *(float4*)(dinv + c * Ndim + j4) = o;
}

// ---------------------------------------------------------------------------
// Hn_bf[i][j] = (i==j ? 0 : H[i][j] * dinv[j]) as bf16  (norm without add).
// ---------------------------------------------------------------------------
__global__ __launch_bounds__(256) void k_scale_cvt(
    const float* __restrict__ H, const float* __restrict__ dinv, bf16* __restrict__ outb)
{
  int c = blockIdx.y;
  size_t idx4 = ((size_t)blockIdx.x * 256 + threadIdx.x) << 2;
  int i = (int)(idx4 >> 11);
  int j = (int)(idx4 & 2047);
  float4 h = *(const float4*)(H + (size_t)c * NN + idx4);
  float4 d = *(const float4*)(dinv + c * Ndim + j);
  float hv[4] = {h.x, h.y, h.z, h.w};
  float dv[4] = {d.x, d.y, d.z, d.w};
  BF4 o;
#pragma unroll
  for (int e = 0; e < 4; ++e)
    o.h[e] = (bf16)((i == j + e) ? 0.f : hv[e] * dv[e]);
  *(uint2*)(outb + (size_t)c * NN + idx4) = o.u;
}

// ---------------------------------------------------------------------------
// XwT[k][i] = (X @ W)^T in bf16 (transposed for readout-GEMM staging).
// ---------------------------------------------------------------------------
__global__ __launch_bounds__(256) void k_xw(
    const float* __restrict__ X, const float* __restrict__ W, bf16* __restrict__ XwT)
{
  int t = threadIdx.x;
  int i = blockIdx.x * 4 + (t >> 6);
  int k = t & 63;
  float s = 0.f;
  for (int j = 0; j < WIN; ++j) s += X[(size_t)i * WIN + j] * W[j * WOUT + k];
  XwT[(size_t)k * Ndim + i] = (bf16)s;
}

// ---------------------------------------------------------------------------
// Readout: out[i, c*64+n] = relu(dinv2[i] * sum_j M[j][i]*Xw[j][n]),
// M = H2 with diagonal := 1 (norm-with-add; column scale hoisted to epilogue).
// A operand staged transposed + diag-fixed + bf16-cvt straight from f32 H2;
// B operand staged (async) from pre-transposed XwT.
// BM=128, BN=64, BK=32; 8 waves = 4(M) x 2(N), 2x2 frags each.
// ---------------------------------------------------------------------------
__global__ __launch_bounds__(256) void k_gemm_out(
    const float* __restrict__ H2, const bf16* __restrict__ XwT,
    const float* __restrict__ dinv, float* __restrict__ out)
{
  const int c  = blockIdx.z;
  const int m0 = blockIdx.y * 128;
  const float* Hc = H2 + (size_t)c * NN;

  __shared__ bf16 sA[128 * LDS_STR];  // sA[m][k] = M[k_glob][m_glob]
  __shared__ bf16 sB[64 * LDS_STR];   // sB[n][k]

  const int tid  = threadIdx.x;
  const int wave = tid >> 5, lane = tid & 31;
  const int lmod = lane & 15, lhi = lane >> 4;
  const int wm = wave & 3;   // 4 waves over M
  const int wn = wave >> 2;  // 2 waves over N

  v8f acc[2][2];
#pragma unroll
  for (int mi = 0; mi < 2; ++mi)
#pragma unroll
    for (int ni = 0; ni < 2; ++ni) acc[mi][ni] = (v8f){0.f,0.f,0.f,0.f,0.f,0.f,0.f,0.f};

  for (int k0 = 0; k0 < Ndim; k0 += 32) {
    __syncthreads();
    // ---- B tile from XwT (row-major copy, one 16B chunk/thread) ----
    {
      int n   = tid >> 2;
      int off = (tid & 3) << 3;
      const bf16* gp = XwT + (size_t)n * Ndim + k0 + off;
      bf16* sp = &sB[n * LDS_STR + off];
      async_copy16(gp, sp);
    }
    // ---- A tile: read rows of H2 (coalesced), scatter transposed + cvt ----
    {
      int kk = tid >> 3;              // 0..31
      int mo = (tid & 7) << 4;        // 0..112 step 16
      int kg = k0 + kk;
      const float* gp = Hc + (size_t)kg * Ndim + m0 + mo;
#pragma unroll
      for (int u = 0; u < 4; ++u) {
        float4 f = *(const float4*)(gp + u * 4);
        float xs[4] = {f.x, f.y, f.z, f.w};
#pragma unroll
        for (int e = 0; e < 4; ++e) {
          int mg = m0 + mo + u * 4 + e;
          float x = (kg == mg) ? 1.0f : xs[e];
          sA[(mo + u * 4 + e) * LDS_STR + kk] = (bf16)x;
        }
      }
    }
    wait_async();
    __syncthreads();

    FragAB aF[2], bF[2];
#pragma unroll
    for (int mi = 0; mi < 2; ++mi) {
      const bf16* p = &sA[(wm * 32 + mi * 16 + lmod) * LDS_STR + lhi * 8];
      aF[mi].q[0] = *(const uint4*)p;
      aF[mi].q[1] = *(const uint4*)(p + 16);
    }
#pragma unroll
    for (int ni = 0; ni < 2; ++ni) {
      const bf16* p = &sB[(wn * 32 + ni * 16 + lmod) * LDS_STR + lhi * 8];
      bF[ni].q[0] = *(const uint4*)p;
      bF[ni].q[1] = *(const uint4*)(p + 16);
    }
#pragma unroll
    for (int mi = 0; mi < 2; ++mi)
#pragma unroll
      for (int ni = 0; ni < 2; ++ni)
        acc[mi][ni] = __builtin_amdgcn_wmma_f32_16x16x32_bf16(
            false, aF[mi].v, false, bF[ni].v, (short)0, acc[mi][ni], false, false);
  }

#pragma unroll
  for (int mi = 0; mi < 2; ++mi)
#pragma unroll
    for (int ni = 0; ni < 2; ++ni) {
      int coln = wn * 32 + ni * 16 + lmod;
#pragma unroll
      for (int v = 0; v < 8; ++v) {
        int row = m0 + wm * 32 + mi * 16 + lhi * 8 + v;
        float r = acc[mi][ni][v] * dinv[c * Ndim + row];
        out[(size_t)row * (Cch * WOUT) + c * WOUT + coln] = fmaxf(r, 0.f);
      }
    }
}

// ---------------------------------------------------------------------------
extern "C" void kernel_launch(void* const* d_in, const int* in_sizes, int n_in,
                              void* d_out, int out_size, void* d_ws, size_t ws_size,
                              hipStream_t stream)
{
  const float* A  = (const float*)d_in[0];   // [5,2048,2048]
  const float* X  = (const float*)d_in[1];   // [2048,256]
  const float* w1 = (const float*)d_in[2];   // [2,5,1,1]
  const float* w2 = (const float*)d_in[3];
  const float* w3 = (const float*)d_in[4];
  const float* W  = (const float*)d_in[5];   // [256,64]
  float* out = (float*)d_out;                // [2048,128]

  // Workspace (~84 MB) with aliasing: H2 reuses H, Hn_bf reuses bT.
  char* ws = (char*)d_ws;
  bf16* a_bf   = (bf16*)ws;  ws += (size_t)Cch * NN * sizeof(bf16);
  bf16* bT_bf  = (bf16*)ws;  ws += (size_t)Cch * NN * sizeof(bf16);  // -> Hn_bf later
  bf16* a1T_bf = (bf16*)ws;  ws += (size_t)Cch * NN * sizeof(bf16);
  float* H     = (float*)ws; ws += (size_t)Cch * NN * sizeof(float); // -> H2 later
  float* dinv1 = (float*)ws; ws += (size_t)Cch * Ndim * sizeof(float);
  float* dinv2 = (float*)ws; ws += (size_t)Cch * Ndim * sizeof(float);
  bf16* XwT    = (bf16*)ws;  ws += (size_t)Ndim * WOUT * sizeof(bf16);

  // 1) softmax-weighted sums: a (row-major), bT, a1T (pre-transposed), bf16
  k_gtconv<<<dim3(Ndim / 32, Ndim / 32), 256, 0, stream>>>(A, w1, w2, w3,
                                                           a_bf, bT_bf, a1T_bf);
  // 2) H = a @ b  (WMMA, f32 accum; B given transposed)
  k_gemm_bf16<<<dim3(16, 16, Cch), 256, 0, stream>>>(a_bf, bT_bf, H, Ndim, Ndim, Ndim);
  // 3) norm(add=false): degrees, then scale+cvt into Hn_bf (aliases bT)
  k_colsum<<<dim3(Ndim / 1024, Cch), 256, 0, stream>>>(H, dinv1, 0);
  k_scale_cvt<<<dim3((unsigned)(NN / 1024), Cch), 256, 0, stream>>>(H, dinv1, bT_bf);
  // 4) H2 = Hn @ a1  (output aliases H)
  k_gemm_bf16<<<dim3(16, 16, Cch), 256, 0, stream>>>(bT_bf, a1T_bf, H, Ndim, Ndim, Ndim);
  // 5) norm(add=true) degrees for readout
  k_colsum<<<dim3(Ndim / 1024, Cch), 256, 0, stream>>>(H, dinv2, 1);
  // 6) XwT = (X @ W)^T in bf16
  k_xw<<<Ndim / 4, 256, 0, stream>>>(X, W, XwT);
  // 7) out = relu(diag-fixed H2^T @ Xw, row-scaled), channels interleaved
  k_gemm_out<<<dim3(1, 16, Cch), 256, 0, stream>>>(H, XwT, dinv2, out);
}
